// GCN1_34282428956966
// MI455X (gfx1250) — matrix operand order repaired and checked by reading, not compiled, not run on hardware
//
#include <hip/hip_runtime.h>

typedef __attribute__((ext_vector_type(16))) _Float16 v16h;
typedef __attribute__((ext_vector_type(8)))  _Float16 v8h;
typedef __attribute__((ext_vector_type(8)))  float    v8f;

#define GCN_N   8192
#define GCN_M   16
#define F_NODE  128
#define HID     512

#define BM 128
#define BN 128
#define BK 32

// ---------------------------------------------------------------------------
// gfx1250 async global->LDS copy (ASYNCcnt path), with safe fallback
// ---------------------------------------------------------------------------
#if __has_builtin(__builtin_amdgcn_global_load_async_to_lds_b128)
#define HAS_ASYNC_LDS 1
#else
#define HAS_ASYNC_LDS 0
#endif

__device__ __forceinline__ void cp16_to_lds(_Float16* l, const _Float16* g) {
#if HAS_ASYNC_LDS
    // builtin prototype (from hipcc diagnostic): first param is
    // '__attribute__((__vector_size__(4*sizeof(int)))) int __device__ *'
    typedef int v4i_ __attribute__((vector_size(4 * sizeof(int))));
    typedef __attribute__((address_space(1))) v4i_ gv4i_;
    typedef __attribute__((address_space(3))) v4i_ lv4i_;
    __builtin_amdgcn_global_load_async_to_lds_b128(
        (gv4i_*)(void*)(g), (lv4i_*)(void*)(l), 0, 0);
#else
    *(v8h*)l = *(const v8h*)g;
#endif
}

__device__ __forceinline__ void async_wait_all() {
#if HAS_ASYNC_LDS
#if __has_builtin(__builtin_amdgcn_s_wait_asynccnt)
    __builtin_amdgcn_s_wait_asynccnt(0);
#else
    asm volatile("s_wait_asynccnt 0" ::: "memory");
#endif
#endif
}

// ---------------------------------------------------------------------------
// dinv[i] = (1 + #valid_neighbors(i))^{-1/2}
// ---------------------------------------------------------------------------
__global__ void k_dinv(const int* __restrict__ edge, float* __restrict__ dinv) {
    int i = blockIdx.x * blockDim.x + threadIdx.x;
    if (i >= GCN_N) return;
    int cnt = 0;
#pragma unroll
    for (int m = 0; m < GCN_M; ++m)
        cnt += (edge[i * GCN_M + m] >= 0) ? 1 : 0;
    dinv[i] = rsqrtf((float)(cnt + 1));
}

// ---------------------------------------------------------------------------
// Convert W (K x HID fp32, row-major) -> Wt (HID x K f16, n-major)
// ---------------------------------------------------------------------------
__global__ void k_wt_convert(const float* __restrict__ W, _Float16* __restrict__ Wt, int K) {
    int t = blockIdx.x * blockDim.x + threadIdx.x;   // t < K*HID
    int k = t / HID;
    int n = t % HID;
    Wt[(size_t)n * K + k] = (_Float16)W[t];
}

// ---------------------------------------------------------------------------
// Convert node features fp32 -> f16 (one time)
// ---------------------------------------------------------------------------
__global__ void k_x_convert(const float* __restrict__ X, _Float16* __restrict__ Xh) {
    int t = blockIdx.x * blockDim.x + threadIdx.x;   // t < N*F_NODE
    Xh[t] = (_Float16)X[t];
}

// ---------------------------------------------------------------------------
// Sparse normalized aggregation (fp32 math, f16 output for the WMMA GEMM):
//   y[i,:] = (f16) dinv[i]*(dinv[i]*x[i,:] + sum_k dinv[j_k]*x[j_k,:])
// grid.x = N, blockDim.x = HID (512)
// ---------------------------------------------------------------------------
__global__ void k_aggregate(const float* __restrict__ x, const int* __restrict__ edge,
                            const float* __restrict__ dinv, _Float16* __restrict__ y) {
    __shared__ int   s_idx[GCN_M];
    __shared__ float s_dj[GCN_M];
    const int i = blockIdx.x;
    const int c = threadIdx.x;
    if (c < GCN_M) {
        int j = edge[i * GCN_M + c];
        s_idx[c] = j;
        s_dj[c]  = (j >= 0) ? dinv[j] : 0.0f;
    }
    __syncthreads();
    const float di = dinv[i];
    float acc = di * x[(size_t)i * HID + c];
#pragma unroll
    for (int m = 0; m < GCN_M; ++m) {
        int j = s_idx[m];
        if (j >= 0) acc += s_dj[m] * x[(size_t)j * HID + c];
    }
    y[(size_t)i * HID + c] = (_Float16)(di * acc);
}

// ---------------------------------------------------------------------------
// C(8192 x 512 fp32) = A(8192 x K, f16) @ Wt^T (Wt is HID x K f16, n-major) + bias (+relu)
// 256 threads = 8 waves; block tile 128x128; wave tile 32x64 (2x4 WMMA 16x16, K-step 32).
// Double-buffered LDS, async global->LDS staging, one barrier per K-step.
// ---------------------------------------------------------------------------
__global__ __launch_bounds__(256) void k_gemm_wmma(
    const _Float16* __restrict__ A, int K,
    const _Float16* __restrict__ Wt,
    const float* __restrict__ bias,
    float* __restrict__ C, int relu)
{
    __shared__ __align__(16) _Float16 sA[2][BM * BK];   // row-major, 128 m x 32 k
    __shared__ __align__(16) _Float16 sB[2][BN * BK];   // n-major,   128 n x 32 k

    const int tid  = threadIdx.x;
    const int lane = tid & 31;
    const int wave = tid >> 5;

    const int m0 = blockIdx.x * BM;
    const int n0 = blockIdx.y * BN;
    const int wm = (wave >> 1) * 32;   // 4 waves along M -> 128
    const int wn = (wave & 1) * 64;    // 2 waves along N -> 128

    // staging: each thread moves 2x16B of A and 2x16B of B per K-step
    const int srow = tid >> 2;          // 0..63
    const int scol = (tid & 3) * 8;     // 0,8,16,24 (halves)

    const _Float16* gA0 = A  + (size_t)(m0 + srow) * K + scol;
    const _Float16* gA1 = A  + (size_t)(m0 + srow + 64) * K + scol;
    const _Float16* gB0 = Wt + (size_t)(n0 + srow) * K + scol;
    const _Float16* gB1 = Wt + (size_t)(n0 + srow + 64) * K + scol;

    // WMMA fragment lane mapping (CDNA5 16-bit 16x32 A / 32x16 B layouts)
    const int rA  = lane & 15;
    const int kA0 = (lane < 16) ? 0 : 8;    // elems 0..7 -> K=kA0.., elems 8..15 -> kA0+16..
    const int nB  = lane & 15;
    const int kB0 = (lane < 16) ? 0 : 8;    // elems 0..7 -> K=kB0.., elems 8..15 -> kB0+16..

    v8f acc[2][4] = {};

    const int KT = K / BK;

    // preload tile 0 into buffer 0
    {
        cp16_to_lds(&sA[0][srow * BK + scol],        gA0);
        cp16_to_lds(&sA[0][(srow + 64) * BK + scol], gA1);
        cp16_to_lds(&sB[0][srow * BK + scol],        gB0);
        cp16_to_lds(&sB[0][(srow + 64) * BK + scol], gB1);
    }
    async_wait_all();
    __syncthreads();

    for (int kt = 0; kt < KT; ++kt) {
        const int cur = kt & 1;
        const int nxt = cur ^ 1;

        // stage next tile into the other buffer (overlaps with WMMA below)
        if (kt + 1 < KT) {
            const int ko = (kt + 1) * BK;
            cp16_to_lds(&sA[nxt][srow * BK + scol],        gA0 + ko);
            cp16_to_lds(&sA[nxt][(srow + 64) * BK + scol], gA1 + ko);
            cp16_to_lds(&sB[nxt][srow * BK + scol],        gB0 + ko);
            cp16_to_lds(&sB[nxt][(srow + 64) * BK + scol], gB1 + ko);
        }

        // fragments from current buffer
        const _Float16* bufA = sA[cur];
        const _Float16* bufB = sB[cur];
        v16h afrag[2], bfrag[4];
#pragma unroll
        for (int i = 0; i < 2; ++i) {
            const _Float16* pa = bufA + (wm + i * 16 + rA) * BK + kA0;
            v8h a0 = *(const v8h*)pa;
            v8h a1 = *(const v8h*)(pa + 16);
#pragma unroll
            for (int e = 0; e < 8; ++e) { afrag[i][e] = a0[e]; afrag[i][8 + e] = a1[e]; }
        }
#pragma unroll
        for (int j = 0; j < 4; ++j) {
            const _Float16* pb = bufB + (wn + j * 16 + nB) * BK + kB0;
            v8h b0 = *(const v8h*)pb;
            v8h b1 = *(const v8h*)(pb + 16);
#pragma unroll
            for (int e = 0; e < 8; ++e) { bfrag[j][e] = b0[e]; bfrag[j][8 + e] = b1[e]; }
        }
#pragma unroll
        for (int i = 0; i < 2; ++i)
#pragma unroll
            for (int j = 0; j < 4; ++j)
                acc[i][j] = __builtin_amdgcn_wmma_f32_16x16x32_f16(
                    false, afrag[i], false, bfrag[j], (short)0, acc[i][j], false, false);

        async_wait_all();
        __syncthreads();
    }

    // epilogue: bias (+relu), fp32 store. C/D layout: N = lane%16, M = r + (lane<16?0:8)
    const int rbase = (lane < 16) ? 0 : 8;
#pragma unroll
    for (int j = 0; j < 4; ++j) {
        const int gn = n0 + wn + j * 16 + nB;
        const float bv = bias[gn];
#pragma unroll
        for (int i = 0; i < 2; ++i) {
#pragma unroll
            for (int r = 0; r < 8; ++r) {
                const int gm = m0 + wm + i * 16 + rbase + r;
                float v = acc[i][j][r] + bv;
                if (relu) v = fmaxf(v, 0.0f);
                C[(size_t)gm * HID + gn] = v;
            }
        }
    }
}

// ---------------------------------------------------------------------------
extern "C" void kernel_launch(void* const* d_in, const int* in_sizes, int n_in,
                              void* d_out, int out_size, void* d_ws, size_t ws_size,
                              hipStream_t stream) {
    const float* node  = (const float*)d_in[0];
    const int*   edge  = (const int*)  d_in[1];
    const float* W_emb = (const float*)d_in[2];
    const float* b_emb = (const float*)d_in[3];
    const float* W1    = (const float*)d_in[4];
    const float* b1    = (const float*)d_in[5];
    const float* W2    = (const float*)d_in[6];
    const float* b2    = (const float*)d_in[7];
    const float* W3    = (const float*)d_in[8];
    const float* b3    = (const float*)d_in[9];
    float* out = (float*)d_out;

    char* ws = (char*)d_ws;
    size_t off = 0;
    float*    dinv  = (float*)(ws + off);    off += (size_t)GCN_N * sizeof(float);        // 32 KB
    _Float16* WtE   = (_Float16*)(ws + off); off += (size_t)F_NODE * HID * 2;             // 128 KB
    _Float16* Wt1   = (_Float16*)(ws + off); off += (size_t)HID * HID * 2;                // 512 KB
    _Float16* Wt2   = (_Float16*)(ws + off); off += (size_t)HID * HID * 2;
    _Float16* Wt3   = (_Float16*)(ws + off); off += (size_t)HID * HID * 2;
    _Float16* Xh    = (_Float16*)(ws + off); off += (size_t)GCN_N * F_NODE * 2;           // 2 MB
    float*    bufX  = (float*)(ws + off);    off += (size_t)GCN_N * HID * sizeof(float);  // 16 MB
    _Float16* bufAh = (_Float16*)(ws + off); off += (size_t)GCN_N * HID * 2;              // 8 MB
    (void)ws_size; (void)in_sizes; (void)n_in; (void)out_size;

    // prep: degrees, f16 transposed weights, f16 node features
    k_dinv<<<GCN_N / 256, 256, 0, stream>>>(edge, dinv);
    k_wt_convert<<<(F_NODE * HID) / 256, 256, 0, stream>>>(W_emb, WtE, F_NODE);
    k_wt_convert<<<(HID * HID) / 256, 256, 0, stream>>>(W1, Wt1, HID);
    k_wt_convert<<<(HID * HID) / 256, 256, 0, stream>>>(W2, Wt2, HID);
    k_wt_convert<<<(HID * HID) / 256, 256, 0, stream>>>(W3, Wt3, HID);
    k_x_convert<<<(GCN_N * F_NODE) / 256, 256, 0, stream>>>(node, Xh);

    dim3 ggrid(GCN_N / BM, HID / BN);   // 64 x 4 blocks

    // embed: X = node @ W_emb + b_emb
    k_gemm_wmma<<<ggrid, 256, 0, stream>>>(Xh, F_NODE, WtE, b_emb, bufX, 0);

    // gcn1: X = relu((adj @ X) @ W1 + b1)
    k_aggregate<<<GCN_N, HID, 0, stream>>>(bufX, edge, dinv, bufAh);
    k_gemm_wmma<<<ggrid, 256, 0, stream>>>(bufAh, HID, Wt1, b1, bufX, 1);

    // gcn2: X = relu((adj @ X) @ W2 + b2)
    k_aggregate<<<GCN_N, HID, 0, stream>>>(bufX, edge, dinv, bufAh);
    k_gemm_wmma<<<ggrid, 256, 0, stream>>>(bufAh, HID, Wt2, b2, bufX, 1);

    // gcn3: out = (adj @ X) @ W3 + b3
    k_aggregate<<<GCN_N, HID, 0, stream>>>(bufX, edge, dinv, bufAh);
    k_gemm_wmma<<<ggrid, 256, 0, stream>>>(bufAh, HID, Wt3, b3, out, 0);
}
